// AttentionMatrix_3384434229669
// MI455X (gfx1250) — compile-verified, matrix operand order checked by reading, not run on hardware
//
#include <hip/hip_runtime.h>
#include <hip/hip_bf16.h>

// Linear attention (AttentionMatrix) for MI455X / gfx1250, wave32 + WMMA bf16.
// S=4096 B=4 D=1024 NH=16 dh=64.  Outputs: [out | k | q], each 16384x1024 f32.

#define S_DIM 4096
#define B_DIM 4
#define D_DIM 1024
#define NHEAD 16
#define DH 64
#define ROWS (S_DIM * B_DIM) // 16384

typedef __attribute__((ext_vector_type(16))) __bf16 v16bf;
typedef __attribute__((ext_vector_type(8))) float v8f;
typedef int async_i4 __attribute__((vector_size(16)));  // b128 payload type

// gfx1250 async global->LDS copies (ASYNCcnt-tracked), if this toolchain has them.
#if defined(__has_builtin)
#if __has_builtin(__builtin_amdgcn_global_load_async_to_lds_b128) && \
    __has_builtin(__builtin_amdgcn_s_wait_asynccnt)
#define HAVE_ASYNC_LDS 1
#endif
#endif

#define AS_GLOBAL(p) ((__attribute__((address_space(1))) async_i4*)(p))
#define AS_LOCAL(p)  ((__attribute__((address_space(3))) async_i4*)(p))

__device__ __forceinline__ __bf16 f2bf(float f) {
  union { float f; unsigned u; } v; v.f = f;
  unsigned r = v.u + 0x7FFFu + ((v.u >> 16) & 1u); // round-to-nearest-even
  union { unsigned short s; __bf16 b; } o; o.s = (unsigned short)(r >> 16);
  return o.b;
}

// A/B fragment for v_wmma_f32_16x16x32_bf16, operand row-major in memory.
// lane%16 selects the row (A: M, B: N); K elements per ISA 7.12.2:
// lanes 0-15 hold K = {0..7, 16..23}, lanes 16-31 hold K = {8..15, 24..31}.
__device__ __forceinline__ v16bf load_frag_bf16(const __bf16* __restrict__ p,
                                                int ld, int lane) {
  const int lr = lane & 15, half = lane >> 4;
  const __bf16* row = p + (size_t)lr * ld + half * 8;
  union { uint4 u[2]; v16bf v; } u;
  u.u[0] = *reinterpret_cast<const uint4*>(row);       // K = 8*half + 0..7
  u.u[1] = *reinterpret_cast<const uint4*>(row + 16);  // K = 16 + 8*half + 0..7
  return u.v;
}

// Fragment sourced from an LDS tile of fixed 32-element row stride.
__device__ __forceinline__ v16bf load_frag_lds(const __bf16* tile, int lane) {
  const int lr = lane & 15, half = lane >> 4;
  const __bf16* row = tile + lr * 32 + half * 8;
  union { uint4 u[2]; v16bf v; } u;
  u.u[0] = *reinterpret_cast<const uint4*>(row);
  u.u[1] = *reinterpret_cast<const uint4*>(row + 16);
  return u.v;
}

__device__ __forceinline__ v8f wmma_bf16(v16bf a, v16bf b, v8f c) {
  return __builtin_amdgcn_wmma_f32_16x16x32_bf16(false, a, false, b,
                                                 (short)0, c, false, false);
}

// ---------------- f32 -> bf16 conversion (weights + activations) ------------
__global__ void cvt_bf16_kernel(const float* __restrict__ src,
                                __bf16* __restrict__ dst, int n) {
  int i = blockIdx.x * blockDim.x + threadIdx.x;
  int stride = gridDim.x * blockDim.x;
  for (; i < n; i += stride) dst[i] = f2bf(src[i]);
}

// ---------------- QKV projection + unitnorm + elu ----------------
// grid (ROWS/16, NHEAD), block 128 (4 waves, each one 16-col tile of the head).
// The shared 16x32 A-tile of h(bf16) is staged into LDS once per K-step via
// async global->LDS copies (double-buffered); B fragments stream from global.
__global__ void __launch_bounds__(128)
qkv_kernel(const __bf16* __restrict__ hbf,
           const __bf16* __restrict__ wqb, const __bf16* __restrict__ wkb,
           const __bf16* __restrict__ wvb,
           const float* __restrict__ bq, const float* __restrict__ bk,
           const float* __restrict__ bv,
           float* __restrict__ kOut, float* __restrict__ qOut,
           __bf16* __restrict__ qbf,   // [b][s][D]
           __bf16* __restrict__ kT,    // [b*NH+h][feat][S]
           __bf16* __restrict__ vT) {  // [b*NH+h][feat][S]
  const int lane = threadIdx.x & 31;
  const int wave = threadIdx.x >> 5;
  const int lr = lane & 15, half = lane >> 4;
  const int rowbase = blockIdx.x * 16;       // rows in native [s*B+b] order
  const int hID = blockIdx.y;
  const int colbase = hID * DH + wave * 16;

  __shared__ __bf16 atile[2][16][32];        // double-buffered h tile
  const __bf16* hA = hbf + (size_t)rowbase * D_DIM;

  // 16 rows x 64B = 64 chunks of 16B; threads 0..63 copy one chunk each.
  const int cr = threadIdx.x >> 2;           // row 0..15
  const int cc = (threadIdx.x & 3) * 8;      // col (elements) 0,8,16,24
  const bool copier = threadIdx.x < 64;

#if HAVE_ASYNC_LDS
#define STAGE_TILE(kk, buf)                                                    \
  if (copier) {                                                                \
    __builtin_amdgcn_global_load_async_to_lds_b128(                            \
        AS_GLOBAL(hA + (size_t)cr * D_DIM + (kk) + cc),                        \
        AS_LOCAL(&atile[buf][cr][cc]), 0, 0);                                  \
  }
#define WAIT_TILE() __builtin_amdgcn_s_wait_asynccnt(0)
#else
#define STAGE_TILE(kk, buf)                                                    \
  if (copier) {                                                                \
    *reinterpret_cast<uint4*>(&atile[buf][cr][cc]) =                           \
        *reinterpret_cast<const uint4*>(hA + (size_t)cr * D_DIM + (kk) + cc);  \
  }
#define WAIT_TILE() ((void)0)
#endif

  v8f cq = {}, ck = {}, cv = {};
  const __bf16* wq = wqb + (size_t)colbase * D_DIM;
  const __bf16* wk = wkb + (size_t)colbase * D_DIM;
  const __bf16* wv = wvb + (size_t)colbase * D_DIM;

  int buf = 0;
  STAGE_TILE(0, 0);
  for (int kk = 0; kk < D_DIM; kk += 32) {
    WAIT_TILE();
    __syncthreads();                          // staged tile visible WG-wide
    if (kk + 32 < D_DIM) STAGE_TILE(kk + 32, buf ^ 1);
    v16bf a = load_frag_lds(&atile[buf][0][0], lane);
    v16bf fq = load_frag_bf16(wq + kk, D_DIM, lane);
    v16bf fk = load_frag_bf16(wk + kk, D_DIM, lane);
    v16bf fv = load_frag_bf16(wv + kk, D_DIM, lane);
    cq = wmma_bf16(a, fq, cq);
    ck = wmma_bf16(a, fk, ck);
    cv = wmma_bf16(a, fv, cv);
    buf ^= 1;
  }

  const int col = colbase + lr;   // C frag: lane -> N(col), vgpr r -> M = r+8*half
  const float biasq = bq[col], biask = bk[col], biasv = bv[col];
#pragma unroll
  for (int r = 0; r < 8; ++r) { cq[r] += biasq; ck[r] += biask; cv[r] += biasv; }

  // Per-row sum of squares across the 64 head columns (4 waves x 16 lanes).
  __shared__ float qss[16], kss[16];
  __syncthreads();
  if (threadIdx.x < 16) qss[threadIdx.x] = 0.0f;
  else if (threadIdx.x < 32) kss[threadIdx.x - 16] = 0.0f;
  __syncthreads();
#pragma unroll
  for (int r = 0; r < 8; ++r) {
    float pq = cq[r] * cq[r];
    float pk = ck[r] * ck[r];
#pragma unroll
    for (int m = 1; m < 16; m <<= 1) {   // butterfly within each 16-lane half
      pq += __shfl_xor(pq, m, 32);
      pk += __shfl_xor(pk, m, 32);
    }
    if (lr == 0) {
      atomicAdd(&qss[r + 8 * half], pq);
      atomicAdd(&kss[r + 8 * half], pk);
    }
  }
  __syncthreads();

  const int feat = wave * 16 + lr;
#pragma unroll
  for (int r = 0; r < 8; ++r) {
    const int m = r + 8 * half;
    const int grow = rowbase + m;          // flattened [s*B+b]
    const int s = grow >> 2, b = grow & 3;
    const int bh = b * NHEAD + hID;
    const float qn = cq[r] * rsqrtf(qss[m]);
    const float kn = ck[r] * rsqrtf(kss[m]);
    const float ke = kn > 0.0f ? kn : (expf(kn) - 1.0f); // elu
    qOut[(size_t)grow * D_DIM + col] = qn;
    kOut[(size_t)grow * D_DIM + col] = ke;
    qbf[((size_t)b * S_DIM + s) * D_DIM + col] = f2bf(qn);
    kT[((size_t)bh * DH + feat) * S_DIM + s] = f2bf(ke);
    vT[((size_t)bh * DH + feat) * S_DIM + s] = f2bf(cv[r]);
  }
}

// ---------------- state accumulation: A[bh] = V^T K  (64x64, K-dim = S) ------
// grid 64 (one WG per (b,h)), block 256 (8 waves split the S dimension).
// Emits A directly in bf16 for the readout GEMM.
__global__ void __launch_bounds__(256)
state_kernel(const __bf16* __restrict__ kT, const __bf16* __restrict__ vT,
             __bf16* __restrict__ Abf) {
  const int lane = threadIdx.x & 31;
  const int wave = threadIdx.x >> 5;
  const int bh = blockIdx.x;
  __shared__ float As[DH * DH];
  for (int i = threadIdx.x; i < DH * DH; i += 256) As[i] = 0.0f;
  __syncthreads();

  v8f acc[4][4] = {};
  const __bf16* vbase = vT + (size_t)bh * DH * S_DIM;
  const __bf16* kbase = kT + (size_t)bh * DH * S_DIM;
  const int s0 = wave * (S_DIM / 8);
  for (int kk = 0; kk < S_DIM / 8; kk += 32) {
    const int sc = s0 + kk;
    v16bf bfr[4];
#pragma unroll
    for (int j = 0; j < 4; ++j)
      bfr[j] = load_frag_bf16(kbase + (size_t)(j * 16) * S_DIM + sc, S_DIM, lane);
#pragma unroll
    for (int i = 0; i < 4; ++i) {
      v16bf afr = load_frag_bf16(vbase + (size_t)(i * 16) * S_DIM + sc, S_DIM, lane);
#pragma unroll
      for (int j = 0; j < 4; ++j) acc[i][j] = wmma_bf16(afr, bfr[j], acc[i][j]);
    }
  }
  const int lr = lane & 15, half = lane >> 4;
#pragma unroll
  for (int i = 0; i < 4; ++i)
#pragma unroll
    for (int j = 0; j < 4; ++j)
#pragma unroll
      for (int r = 0; r < 8; ++r)
        atomicAdd(&As[(i * 16 + r + 8 * half) * DH + j * 16 + lr], acc[i][j][r]);
  __syncthreads();
  __bf16* Ao = Abf + (size_t)bh * DH * DH;
  for (int i = threadIdx.x; i < DH * DH; i += 256) Ao[i] = f2bf(As[i]);
}

// ---------------- readout: oat[row, vi] = sum_qi q[row,qi] * A[vi,qi] --------
// grid (ROWS/16, NHEAD), rows in [b][s] order so each 16-row tile has fixed b.
__global__ void __launch_bounds__(128)
readout_kernel(const __bf16* __restrict__ qbf, const __bf16* __restrict__ Abf,
               __bf16* __restrict__ oat) {
  const int lane = threadIdx.x & 31;
  const int wave = threadIdx.x >> 5;
  const int lr = lane & 15, half = lane >> 4;
  const int rowbase = blockIdx.x * 16;
  const int hID = blockIdx.y;
  const int b = rowbase >> 12;               // 4096 rows per batch
  const __bf16* Ab = Abf + (size_t)(b * NHEAD + hID) * DH * DH;
  const __bf16* qA = qbf + (size_t)rowbase * D_DIM + hID * DH;

  v8f c = {};
  for (int kk = 0; kk < DH; kk += 32) {
    v16bf a = load_frag_bf16(qA + kk, D_DIM, lane);
    v16bf bb = load_frag_bf16(Ab + (size_t)(wave * 16) * DH + kk, DH, lane);
    c = wmma_bf16(a, bb, c);
  }
  const int col = hID * DH + wave * 16 + lr;
#pragma unroll
  for (int r = 0; r < 8; ++r) {
    const int row = rowbase + r + 8 * half;
    oat[(size_t)row * D_DIM + col] = f2bf(c[r]);
  }
}

// ---------------- output projection: out = oat @ Wo^T + bo -------------------
// grid (ROWS/16, D/64); rows in [b][s] order, scattered back to [s][b] on store
__global__ void __launch_bounds__(128)
oproj_kernel(const __bf16* __restrict__ oat, const __bf16* __restrict__ wob,
             const float* __restrict__ bo, float* __restrict__ out) {
  const int lane = threadIdx.x & 31;
  const int wave = threadIdx.x >> 5;
  const int lr = lane & 15, half = lane >> 4;
  const int rowbase = blockIdx.x * 16;
  const int colbase = blockIdx.y * 64 + wave * 16;

  v8f c = {};
  const __bf16* aA = oat + (size_t)rowbase * D_DIM;
  const __bf16* wB = wob + (size_t)colbase * D_DIM;
  for (int kk = 0; kk < D_DIM; kk += 32) {
    v16bf a = load_frag_bf16(aA + kk, D_DIM, lane);
    v16bf bb = load_frag_bf16(wB + kk, D_DIM, lane);
    c = wmma_bf16(a, bb, c);
  }
  const int col = colbase + lr;
  const float bias = bo[col];
#pragma unroll
  for (int r = 0; r < 8; ++r) {
    const int rowp = rowbase + r + 8 * half;   // [b][s] order
    const int b = rowp >> 12, s = rowp & 4095;
    out[((size_t)s * B_DIM + b) * D_DIM + col] = c[r] + bias;
  }
}

extern "C" void kernel_launch(void* const* d_in, const int* in_sizes, int n_in,
                              void* d_out, int out_size, void* d_ws, size_t ws_size,
                              hipStream_t stream) {
  (void)in_sizes; (void)n_in; (void)out_size; (void)ws_size;
  const float* h  = (const float*)d_in[0];
  const float* Wq = (const float*)d_in[1];
  const float* bq = (const float*)d_in[2];
  const float* Wk = (const float*)d_in[3];
  const float* bk = (const float*)d_in[4];
  const float* Wv = (const float*)d_in[5];
  const float* bv = (const float*)d_in[6];
  const float* Wo = (const float*)d_in[7];
  const float* bo = (const float*)d_in[8];

  float* out  = (float*)d_out;                 // [S,B,D]
  float* kOut = out + (size_t)ROWS * D_DIM;    // [S,B,NH,dh] == [S,B,D]
  float* qOut = kOut + (size_t)ROWS * D_DIM;

  const size_t DD = (size_t)D_DIM * D_DIM;     // 1M
  const size_t RD = (size_t)ROWS * D_DIM;      // 16M
  __bf16* wsb = (__bf16*)d_ws;
  __bf16* wqb = wsb;
  __bf16* wkb = wsb + DD;
  __bf16* wvb = wsb + 2 * DD;
  __bf16* wob = wsb + 3 * DD;
  __bf16* hbf = wsb + 4 * DD;                  // h in bf16, native row order
  __bf16* qbf = hbf + RD;
  __bf16* kT  = qbf + RD;
  __bf16* vT  = kT + RD;
  __bf16* oat = vT + RD;
  __bf16* Abf = oat + RD;                      // 64 heads * 64 * 64 bf16

  cvt_bf16_kernel<<<1024, 256, 0, stream>>>(Wq, wqb, (int)DD);
  cvt_bf16_kernel<<<1024, 256, 0, stream>>>(Wk, wkb, (int)DD);
  cvt_bf16_kernel<<<1024, 256, 0, stream>>>(Wv, wvb, (int)DD);
  cvt_bf16_kernel<<<1024, 256, 0, stream>>>(Wo, wob, (int)DD);
  cvt_bf16_kernel<<<4096, 256, 0, stream>>>(h, hbf, (int)RD);

  qkv_kernel<<<dim3(ROWS / 16, NHEAD), 128, 0, stream>>>(
      hbf, wqb, wkb, wvb, bq, bk, bv, kOut, qOut, qbf, kT, vT);
  state_kernel<<<B_DIM * NHEAD, 256, 0, stream>>>(kT, vT, Abf);
  readout_kernel<<<dim3(ROWS / 16, NHEAD), 128, 0, stream>>>(qbf, Abf, oat);
  oproj_kernel<<<dim3(ROWS / 16, D_DIM / 64), 128, 0, stream>>>(oat, wob, bo, out);
}
